// QMLPResidual_73847667687982
// MI455X (gfx1250) — 8-bit matrix operand order repaired and checked by reading, compiled, not run on hardware
//
#include <hip/hip_runtime.h>
#include <hip/hip_bf16.h>
#include <cstdint>
#include <cstddef>


typedef __attribute__((ext_vector_type(8))) int v8i;

#ifndef __has_builtin
#define __has_builtin(x) 0
#endif

#if __has_builtin(__builtin_amdgcn_global_load_async_to_lds_b128)
#define HAVE_ASYNC_LDS 1
#else
#define HAVE_ASYNC_LDS 0
#endif

typedef int v4i_b __attribute__((vector_size(16)));
typedef __attribute__((address_space(1))) v4i_b as1_v4i;
typedef __attribute__((address_space(3))) v4i_b as3_v4i;

__device__ __forceinline__ void cp16_g2l(void* lds_dst, const void* gsrc) {
#if HAVE_ASYNC_LDS
  __builtin_amdgcn_global_load_async_to_lds_b128(
      (as1_v4i*)const_cast<void*>(gsrc), (as3_v4i*)lds_dst, 0, 0);
#else
  *(int4*)lds_dst = *(const int4*)gsrc;
#endif
}

__device__ __forceinline__ void wait_g2l() {
#if HAVE_ASYNC_LDS
#if __has_builtin(__builtin_amdgcn_s_wait_asynccnt)
  __builtin_amdgcn_s_wait_asynccnt(0);
#else
  asm volatile("s_wait_asynccnt 0" ::: "memory");
#endif
#endif
}

__device__ __forceinline__ float qsc(float amax) {
  return fmaxf(amax * (1.0f / 127.0f), 1e-8f);
}
__device__ __forceinline__ int qv(float v, float inv_s) {
  float r = rintf(v * inv_s);
  r = fminf(fmaxf(r, -127.0f), 127.0f);
  return (int)r;
}

__global__ __launch_bounds__(256) void quant_groups_f32(
    const float* __restrict__ src, int8_t* __restrict__ q,
    float* __restrict__ s, int lg, int K, long totalGroups) {
  const long gid = (long)blockIdx.x * 256 + threadIdx.x;
  if (gid >= totalGroups) return;
  const long row = gid >> lg;
  const int g = (int)(gid & ((1 << lg) - 1));
  const float* p = src + row * (long)K + (long)g * 32;
  float v[32];
  float amax = 0.0f;
#pragma unroll
  for (int j = 0; j < 32; j += 4) {
    float4 t = *(const float4*)(p + j);
    v[j] = t.x; v[j + 1] = t.y; v[j + 2] = t.z; v[j + 3] = t.w;
  }
#pragma unroll
  for (int j = 0; j < 32; j++) amax = fmaxf(amax, fabsf(v[j]));
  const float sc = qsc(amax), inv = 1.0f / sc;
  s[gid] = sc;
  int* qp = (int*)(q + row * (long)K + (long)g * 32);
#pragma unroll
  for (int j = 0; j < 32; j += 4) {
    int p0 = qv(v[j], inv) & 255, p1 = qv(v[j + 1], inv) & 255;
    int p2 = qv(v[j + 2], inv) & 255, p3 = qv(v[j + 3], inv) & 255;
    qp[j >> 2] = p0 | (p1 << 8) | (p2 << 16) | (p3 << 24);
  }
}

__global__ __launch_bounds__(256) void ln_quant_kernel(
    const int* __restrict__ qx, const float* __restrict__ sx,
    const float* __restrict__ mean, const float* __restrict__ rstd,
    const float* __restrict__ lnw, int8_t* __restrict__ qh,
    float* __restrict__ sh) {
  const long gid = (long)blockIdx.x * 256 + threadIdx.x;
  const long row = gid >> 6;
  const int g = (int)(gid & 63);
  const float sxv = sx[gid];
  const float m = mean[row], r = rstd[row];
  const int* p = qx + row * 2048 + g * 32;
  float v[32];
  float amax = 0.0f;
#pragma unroll
  for (int j = 0; j < 32; j += 4) {
    int4 t = *(const int4*)(p + j);
    v[j] = (float)t.x; v[j + 1] = (float)t.y;
    v[j + 2] = (float)t.z; v[j + 3] = (float)t.w;
  }
#pragma unroll
  for (int j = 0; j < 32; j++) {
    const float xf = v[j] * sxv;
    const float h = (xf - m) * r * lnw[g * 32 + j];
    v[j] = h;
    amax = fmaxf(amax, fabsf(h));
  }
  const float sc = qsc(amax), inv = 1.0f / sc;
  sh[gid] = sc;
  int* qp = (int*)(qh + row * 2048 + g * 32);
#pragma unroll
  for (int j = 0; j < 32; j += 4) {
    int p0 = qv(v[j], inv) & 255, p1 = qv(v[j + 1], inv) & 255;
    int p2 = qv(v[j + 2], inv) & 255, p3 = qv(v[j + 3], inv) & 255;
    qp[j >> 2] = p0 | (p1 << 8) | (p2 << 16) | (p3 << 24);
  }
}

template <int MODE>
__global__ __launch_bounds__(256) void qgemm_kernel(
    const int8_t* __restrict__ A, const float* __restrict__ sA,
    const int8_t* __restrict__ B, const float* __restrict__ sB, int K,
    int ldOut, int8_t* __restrict__ qOut, float* __restrict__ sOut,
    const float* __restrict__ xres, float* __restrict__ outf) {
  __shared__ char smem[32768];

  const int tid = threadIdx.x;
  const int lane = tid & 31;
  const int w = tid >> 5;
  const int wm = w & 3;
  const int wn = w >> 2;
  const int half = lane >> 4;
  const int l16 = lane & 15;

  const int rowBase = blockIdx.y * 64;
  const int colBase = blockIdx.x * 128;
  const int gpr = K >> 5;

  float acc[4][8];
#pragma unroll
  for (int t = 0; t < 4; t++)
#pragma unroll
    for (int i = 0; i < 8; i++) acc[t][i] = 0.0f;

  const int arow = tid >> 2, agq = tid & 3;

  for (int k0 = 0; k0 < K; k0 += 128) {
    __syncthreads();
    {
      const int8_t* g = A + (size_t)(rowBase + arow) * K + k0 + agq * 32;
      char* l = smem + agq * 2048 + arow * 32;
      cp16_g2l(l, g);
      cp16_g2l(l + 16, g + 16);
    }
#pragma unroll
    for (int it = 0; it < 2; ++it) {
      const int c = tid + it * 256;
      const int col = c & 127, gq2 = c >> 7;
      const int8_t* g = B + (size_t)(colBase + col) * K + k0 + gq2 * 32;
      char* l = smem + 8192 + gq2 * 4096 + col * 32;
      cp16_g2l(l, g);
      cp16_g2l(l + 16, g + 16);
    }
    if (k0 + 128 < K)
      __builtin_prefetch(B + (size_t)(colBase + (tid & 127)) * K + k0 + 128, 0, 1);
    wait_g2l();
    __syncthreads();

#pragma unroll
    for (int q = 0; q < 4; q++) {
      const int gidx = (k0 >> 5) + q;
      const char* ab = smem + q * 2048 + (wm * 16 + l16) * 32 + half * 8;
      const int2 a0 = *(const int2*)ab;
      const int2 a1 = *(const int2*)(ab + 16);
      v8i af;
      af[0] = a0.x; af[1] = a0.y; af[2] = a1.x; af[3] = a1.y;
      af[4] = 0; af[5] = 0; af[6] = 0; af[7] = 0;

      float sAv[8];
#pragma unroll
      for (int i = 0; i < 8; i++)
        sAv[i] = sA[(size_t)(rowBase + wm * 16 + i + 8 * half) * gpr + gidx];

#pragma unroll
      for (int t = 0; t < 4; t++) {
        const int col = wn * 64 + t * 16 + l16;
        const char* bb = smem + 8192 + q * 4096 + col * 32 + half * 8;
        const int2 b0 = *(const int2*)bb;
        const int2 b1 = *(const int2*)(bb + 16);
        v8i bf;
        bf[0] = b0.x; bf[1] = b0.y; bf[2] = b1.x; bf[3] = b1.y;
        bf[4] = 0; bf[5] = 0; bf[6] = 0; bf[7] = 0;
        v8i cz = {0, 0, 0, 0, 0, 0, 0, 0};
        v8i d = __builtin_amdgcn_wmma_i32_16x16x64_iu8(true, af, true, bf, cz,
                                                       false, false);
        const float sBv = sB[(size_t)(colBase + col) * gpr + gidx];
#pragma unroll
        for (int i = 0; i < 8; i++)
          acc[t][i] += (float)d[i] * (sAv[i] * sBv);
      }
    }
  }

  __syncthreads();
  float* st = (float*)smem;
#pragma unroll
  for (int t = 0; t < 4; t++) {
    const int c = wn * 64 + t * 16 + l16;
#pragma unroll
    for (int i = 0; i < 8; i++) {
      const int r = wm * 16 + i + 8 * half;
      st[r * 128 + c] = acc[t][i];
    }
  }
  __syncthreads();

  {
    const int r = tid >> 2, gq = tid & 3;
    float v[32];
    float amax = 0.0f;
#pragma unroll
    for (int j = 0; j < 32; j += 4) {
      float4 t4 = *(const float4*)(st + r * 128 + gq * 32 + j);
      v[j] = t4.x; v[j + 1] = t4.y; v[j + 2] = t4.z; v[j + 3] = t4.w;
    }
#pragma unroll
    for (int j = 0; j < 32; j++) amax = fmaxf(amax, fabsf(v[j]));
    const float s1 = qsc(amax), inv1 = 1.0f / s1;
    const size_t orow = (size_t)(rowBase + r);

    if (MODE == 0) {
      float gbuf[32];
      float amax2 = 0.0f;
#pragma unroll
      for (int j = 0; j < 32; j++) {
        const float dq = (float)qv(v[j], inv1) * s1;
        const float gg = 0.5f * dq * (1.0f + erff(dq * 0.70710678118654752f));
        gbuf[j] = gg;
        amax2 = fmaxf(amax2, fabsf(gg));
      }
      const float s2 = qsc(amax2), inv2 = 1.0f / s2;
      int* qp = (int*)(qOut + orow * ldOut + colBase + gq * 32);
#pragma unroll
      for (int j = 0; j < 32; j += 4) {
        int p0 = qv(gbuf[j], inv2) & 255, p1 = qv(gbuf[j + 1], inv2) & 255;
        int p2 = qv(gbuf[j + 2], inv2) & 255, p3 = qv(gbuf[j + 3], inv2) & 255;
        qp[j >> 2] = p0 | (p1 << 8) | (p2 << 16) | (p3 << 24);
      }
      sOut[orow * (ldOut >> 5) + (colBase >> 5) + gq] = s2;
    } else {
      float* op = outf + orow * ldOut + colBase + gq * 32;
      const float* xp = xres + orow * ldOut + colBase + gq * 32;
#pragma unroll
      for (int j = 0; j < 32; j++) {
        const float dq = (float)qv(v[j], inv1) * s1;
        op[j] = xp[j] + dq;
      }
    }
  }
}

__global__ __launch_bounds__(256) void finalize_kernel(
    const float* __restrict__ outf, __hip_bfloat16* __restrict__ obf,
    int8_t* __restrict__ qo, float* __restrict__ so,
    float* __restrict__ nmean, float* __restrict__ nrstd) {
  __shared__ float rbuf[2048];
  __shared__ float rs[256];
  __shared__ float rs2[256];
  const int row = blockIdx.x, tid = threadIdx.x;
  const float* p = outf + (size_t)row * 2048 + tid * 8;
  float v[8];
#pragma unroll
  for (int j = 0; j < 8; j += 4) {
    float4 t4 = *(const float4*)(p + j);
    v[j] = t4.x; v[j + 1] = t4.y; v[j + 2] = t4.z; v[j + 3] = t4.w;
  }
  float s = 0.0f, s2 = 0.0f;
#pragma unroll
  for (int j = 0; j < 8; j++) {
    s += v[j];
    s2 += v[j] * v[j];
    rbuf[tid * 8 + j] = v[j];
    obf[(size_t)row * 2048 + tid * 8 + j] = __float2bfloat16(v[j]);
  }
  rs[tid] = s;
  rs2[tid] = s2;
  __syncthreads();
  for (int off = 128; off > 0; off >>= 1) {
    if (tid < off) {
      rs[tid] += rs[tid + off];
      rs2[tid] += rs2[tid + off];
    }
    __syncthreads();
  }
  const float mean = rs[0] * (1.0f / 2048.0f);
  const float var = rs2[0] * (1.0f / 2048.0f) - mean * mean;
  const float rstd = 1.0f / sqrtf(var + 1e-6f);
  if (tid == 0) {
    nmean[row] = mean;
    nrstd[row] = rstd;
  }
  if (tid < 64) {
    float amax = 0.0f;
#pragma unroll
    for (int j = 0; j < 32; j++) amax = fmaxf(amax, fabsf(rbuf[tid * 32 + j]));
    const float sc = qsc(amax), inv = 1.0f / sc;
    so[(size_t)row * 64 + tid] = sc;
    int* qp = (int*)(qo + (size_t)row * 2048 + tid * 32);
#pragma unroll
    for (int j = 0; j < 32; j += 4) {
      int p0 = qv(rbuf[tid * 32 + j], inv) & 255;
      int p1 = qv(rbuf[tid * 32 + j + 1], inv) & 255;
      int p2 = qv(rbuf[tid * 32 + j + 2], inv) & 255;
      int p3 = qv(rbuf[tid * 32 + j + 3], inv) & 255;
      qp[j >> 2] = p0 | (p1 << 8) | (p2 << 16) | (p3 << 24);
    }
  }
}

extern "C" void kernel_launch(void* const* d_in, const int* in_sizes, int n_in,
                              void* d_out, int out_size, void* d_ws,
                              size_t ws_size, hipStream_t stream) {
  (void)in_sizes; (void)n_in; (void)out_size; (void)ws_size;
  const int N = 4096, D = 2048, H = 8192;

  const float* x = (const float*)d_in[0];
  const int* qx = (const int*)d_in[1];
  const float* sx = (const float*)d_in[2];
  const float* mean = (const float*)d_in[3];
  const float* rstd = (const float*)d_in[4];
  const float* w1 = (const float*)d_in[5];
  const float* w2 = (const float*)d_in[6];
  const float* lnw = (const float*)d_in[7];

  char* ws = (char*)d_ws;
  size_t off = 0;
  auto alloc = [&](size_t bytes) -> char* {
    char* p = ws + off;
    off += (bytes + 255) & ~(size_t)255;
    return p;
  };
  int8_t* qw1 = (int8_t*)alloc((size_t)H * D);
  float* sw1 = (float*)alloc((size_t)H * (D / 32) * 4);
  int8_t* qw2 = (int8_t*)alloc((size_t)D * H);
  float* sw2 = (float*)alloc((size_t)D * (H / 32) * 4);
  int8_t* qh = (int8_t*)alloc((size_t)N * D);
  float* shs = (float*)alloc((size_t)N * (D / 32) * 4);
  int8_t* qg = (int8_t*)alloc((size_t)N * H);
  float* sg = (float*)alloc((size_t)N * (H / 32) * 4);
  float* outf = (float*)alloc((size_t)N * D * 4);

  char* ob = (char*)d_out;
  __hip_bfloat16* obf = (__hip_bfloat16*)ob;
  int8_t* qo = (int8_t*)(ob + (size_t)N * D * 2);
  float* so = (float*)(ob + (size_t)N * D * 2 + (size_t)N * D);
  float* nmean = so + (size_t)N * (D / 32);
  float* nrstd = nmean + N;

  quant_groups_f32<<<2048, 256, 0, stream>>>(w1, qw1, sw1, 6, D,
                                             (long)H * (D / 32));
  quant_groups_f32<<<2048, 256, 0, stream>>>(w2, qw2, sw2, 8, H,
                                             (long)D * (H / 32));
  ln_quant_kernel<<<1024, 256, 0, stream>>>(qx, sx, mean, rstd, lnw, qh, shs);
  qgemm_kernel<0><<<dim3(H / 128, N / 64), 256, 0, stream>>>(
      qh, shs, qw1, sw1, D, H, qg, sg, nullptr, nullptr);
  qgemm_kernel<1><<<dim3(D / 128, N / 64), 256, 0, stream>>>(
      qg, sg, qw2, sw2, H, D, nullptr, nullptr, x, outf);
  finalize_kernel<<<N, 256, 0, stream>>>(outf, obf, qo, so, nmean, nrstd);
}
